// SelfAttention3D_3917010174535
// MI455X (gfx1250) — matrix operand order repaired and checked by reading, not compiled, not run on hardware
//
#include <hip/hip_runtime.h>

typedef __bf16 bf16_t;
typedef __attribute__((ext_vector_type(16))) __bf16        v16bf;
typedef __attribute__((ext_vector_type(8)))  float         v8f;
typedef __attribute__((ext_vector_type(4)))  unsigned int  v4u;

namespace {
constexpr int kB  = 4;
constexpr int kC  = 256;
constexpr int kN  = 4096;    // 16*16*16 voxels
constexpr int kI  = 32;      // INNER
constexpr int kMC = kN / 32; // 128 m-chunks of 32
}

// Detect gfx1250 async-to-LDS builtins (present on newer toolchains).
#if __has_builtin(__builtin_amdgcn_global_load_async_to_lds_b128) && \
    __has_builtin(__builtin_amdgcn_s_wait_asynccnt)
#define USE_ASYNC_LDS 1
#endif

#if defined(USE_ASYNC_LDS)
// Payload-typed pointers per the builtin's signature: b128 => <4 x i32>.
typedef int gv4i __attribute__((vector_size(16)));
typedef __attribute__((address_space(3))) gv4i as3_gv4i;
static __device__ __forceinline__ gv4i* gsrc_cast(const void* p) {
  return (gv4i*)(unsigned long long)(size_t)p;   // generic pointer, v4i-typed
}
static __device__ __forceinline__ as3_gv4i* lds_cast(void* p) {
  // LDS generic addresses keep the LDS offset in the low 32 bits (ISA 10.2).
  return (as3_gv4i*)(unsigned int)(size_t)p;
}
#endif

// Truncating f32->bf16 pair pack: one v_perm_b32 (bytes {hi.b3,hi.b2,lo.b3,lo.b2}).
static __device__ __forceinline__ unsigned int pack2(float lo, float hi) {
  return __builtin_amdgcn_perm(__builtin_bit_cast(unsigned int, hi),
                               __builtin_bit_cast(unsigned int, lo),
                               0x07060302u);
}
static __device__ __forceinline__ unsigned short f2bf(float f) {
  return (unsigned short)(__builtin_bit_cast(unsigned int, f) >> 16);
}
static __device__ __forceinline__ v8f wmma_bf16(v16bf a, v16bf b, v8f c) {
  return __builtin_amdgcn_wmma_f32_16x16x32_bf16(false, a, false, b, (short)0, c,
                                                 false, false);
}

// ---------------------------------------------------------------------------
// Projection: q = Wq x + bq, k = Wk x + bk, v = Wv x + bv  (bf16 WMMA, f32 acc)
//   qT,kT : [B][N][32] bf16 (rows load directly as WMMA A (q) / B (k) operands)
//   vB    : [B][mc][t][lane][16] bf16 — V pre-swizzled into the per-lane WMMA
//           B-operand layout for the PV matmul (1 KB per 32x16 tile)
// ---------------------------------------------------------------------------
__global__ void __launch_bounds__(32)
proj_kernel(const float* __restrict__ x,
            const float* __restrict__ Wq, const float* __restrict__ bq,
            const float* __restrict__ Wk, const float* __restrict__ bk,
            const float* __restrict__ Wv, const float* __restrict__ bv,
            bf16_t* __restrict__ qT, bf16_t* __restrict__ kT,
            bf16_t* __restrict__ vB)
{
  __shared__ unsigned short vstage[32][18];   // [m][c] bf16, padded pitch
  const int lane  = threadIdx.x;
  const int l16   = lane & 15;
  const int lhalf = lane >> 4;
  const int wid   = blockIdx.x;
  const int b     = wid >> 7;                 // / (kN/32)
  const int mc    = wid & (kMC - 1);
  const int n0    = mc * 32;

  const float* xb = x + (size_t)b * kC * kN;

  // Stage x as B operands (32 c-rows x 16 n-cols) for 8 K-steps, both halves.
  // B operands here use contiguous k order: lanes 0-15 col n=lane hold K 0..15; lanes 16-31 K 16..31.
  v16bf xs0[8], xs1[8];
  #pragma unroll
  for (int ks = 0; ks < 8; ++ks) {
    const int cbase = ks * 32 + lhalf * 16;
    float a0[16], a1[16];
    #pragma unroll
    for (int j = 0; j < 16; ++j) {
      const size_t rowoff = (size_t)(cbase + j) * kN;
      a0[j] = xb[rowoff + n0 + l16];
      a1[j] = xb[rowoff + n0 + 16 + l16];
    }
    union { v16bf v; unsigned int u[8]; } t0, t1;
    #pragma unroll
    for (int j = 0; j < 8; ++j) {
      t0.u[j] = pack2(a0[2 * j], a0[2 * j + 1]);
      t1.u[j] = pack2(a1[2 * j], a1[2 * j + 1]);
    }
    xs0[ks] = t0.v; xs1[ks] = t1.v;
  }

  // A operand from row-major weights: row o=l16+obase, K 16*lhalf..+15
  auto loadA = [&](const float* __restrict__ W, int obase, int ks) -> v16bf {
    const float* r = W + (size_t)(obase + l16) * kC + ks * 32 + lhalf * 16;
    union { v16bf v; unsigned int u[8]; } t;
    #pragma unroll
    for (int j = 0; j < 4; ++j) t.u[j]     = pack2(r[2 * j],      r[2 * j + 1]);
    #pragma unroll
    for (int j = 0; j < 4; ++j) t.u[4 + j] = pack2(r[8 + 2 * j], r[8 + 2 * j + 1]);
    return t.v;
  };

  auto gemm_pair = [&](const float* __restrict__ W, const float* __restrict__ bias,
                       int obase, v8f& o0, v8f& o1) {
    v8f c0 = {}; v8f c1 = {};
    #pragma unroll
    for (int ks = 0; ks < 8; ++ks) {
      v16bf a = loadA(W, obase, ks);
      c0 = wmma_bf16(a, xs0[ks], c0);
      c1 = wmma_bf16(a, xs1[ks], c1);
    }
    #pragma unroll
    for (int r = 0; r < 8; ++r) {
      const float bb = bias[obase + lhalf * 8 + r];
      c0[r] += bb; c1[r] += bb;
    }
    o0 = c0; o1 = c1;
  };

  // q/k: store transposed [n][32]; D layout gives each lane 8 consecutive o's.
  auto writeQK = [&](bf16_t* __restrict__ dst, int obase, v8f c0, v8f c1) {
    #pragma unroll
    for (int h = 0; h < 2; ++h) {
      const v8f c = h ? c1 : c0;
      const int n = n0 + h * 16 + l16;
      v4u u = { pack2(c[0], c[1]), pack2(c[2], c[3]),
                pack2(c[4], c[5]), pack2(c[6], c[7]) };
      *(v4u*)(dst + (size_t)n * kI + obase + lhalf * 8) = u;
    }
  };

  bf16_t* qTb = qT + (size_t)b * kN * kI;
  bf16_t* kTb = kT + (size_t)b * kN * kI;
  {
    v8f c0, c1;
    gemm_pair(Wq, bq, 0,  c0, c1); writeQK(qTb, 0,  c0, c1);
    gemm_pair(Wq, bq, 16, c0, c1); writeQK(qTb, 16, c0, c1);
    gemm_pair(Wk, bk, 0,  c0, c1); writeQK(kTb, 0,  c0, c1);
    gemm_pair(Wk, bk, 16, c0, c1); writeQK(kTb, 16, c0, c1);
  }

  // V: 16 o-tiles; transpose D-layout -> B-operand layout through LDS.
  for (int t16 = 0; t16 < 16; ++t16) {
    v8f c0, c1;
    gemm_pair(Wv, bv, t16 * 16, c0, c1);
    #pragma unroll
    for (int r = 0; r < 8; ++r) {
      vstage[l16][lhalf * 8 + r]      = f2bf(c0[r]);   // m=l16    , c=row
      vstage[16 + l16][lhalf * 8 + r] = f2bf(c1[r]);   // m=16+l16 , c=row
    }
    asm volatile("s_wait_dscnt 0" ::: "memory");
    // lane's 16 B-operand halves: m = lhalf*16 + j, c = l16
    unsigned int u[8];
    #pragma unroll
    for (int j = 0; j < 8; ++j) {
      const unsigned int lo = vstage[lhalf * 16 + 2 * j][l16];
      const unsigned int hi = vstage[lhalf * 16 + 2 * j + 1][l16];
      u[j] = lo | (hi << 16);
    }
    bf16_t* dst = vB + ((((size_t)b * kMC + mc) * 16 + t16) * 32 + lane) * 16;
    v4u a = { u[0], u[1], u[2], u[3] };
    v4u c = { u[4], u[5], u[6], u[7] };
    *(v4u*)(dst)     = a;
    *(v4u*)(dst + 8) = c;
  }
}

// ---------------------------------------------------------------------------
// Flash attention: 4 waves/block, one 16-row n-tile per wave. All 4 waves walk
// the SAME m-chunks, so the 16 KB V chunk is staged into LDS once per block
// via GLOBAL_LOAD_ASYNC_TO_LDS_B128 (double-buffered, ASYNCcnt + barriers),
// cutting V L2 traffic 4x. Per 32-wide m chunk: 2 QK WMMAs -> online softmax
// (shfl_xor row reductions) -> P staged through per-wave LDS (D->A layout)
// -> 16 PV WMMAs. Epilogue: out = gamma * (O / l) + x.
// ---------------------------------------------------------------------------
__global__ void __launch_bounds__(128)
attn_kernel(const float* __restrict__ x,
            const bf16_t* __restrict__ qT, const bf16_t* __restrict__ kT,
            const bf16_t* __restrict__ vB,
            const float* __restrict__ gamma, float* __restrict__ out)
{
  __shared__ unsigned short pstage[4][16][40];   // per-wave [n][m], padded pitch
#if defined(USE_ASYNC_LDS)
  __shared__ alignas(16) char vbuf[2][16384];    // double-buffered V chunk
#endif
  const int tid   = threadIdx.x;
  const int lane  = tid & 31;
  const int w     = tid >> 5;
  const int l16   = lane & 15;
  const int lhalf = lane >> 4;
  const int rt    = blockIdx.x * 4 + w;
  const int b     = rt >> 8;                     // kN/16 = 256 row tiles/batch
  const int nbase = (rt & 255) * 16;

  const bf16_t* qb = qT + (size_t)b * kN * kI;
  const bf16_t* kb = kT + (size_t)b * kN * kI;
  const bf16_t* vb = vB + (size_t)b * kMC * 16 * 512;

  // Q tile as WMMA A operand: row=l16, K 16*lhalf..+15
  union { v16bf v; v4u u[2]; } aq;
  {
    const bf16_t* qr = qb + (size_t)(nbase + l16) * kI + lhalf * 16;
    aq.u[0] = *(const v4u*)(qr);
    aq.u[1] = *(const v4u*)(qr + 8);
  }

  v8f acc[16];
  #pragma unroll
  for (int t = 0; t < 16; ++t) acc[t] = (v8f){};
  float m_s[8], l_s[8];
  #pragma unroll
  for (int r = 0; r < 8; ++r) { m_s[r] = -__builtin_inff(); l_s[r] = 0.f; }

#if defined(USE_ASYNC_LDS)
  // Prologue: async-stage chunk 0 (1024 16B units; 8 per thread).
  {
    const bf16_t* src = vb + (size_t)tid * 8;
    char* dst = vbuf[0] + tid * 16;
    #pragma unroll
    for (int i = 0; i < 8; ++i)
      __builtin_amdgcn_global_load_async_to_lds_b128(
          gsrc_cast(src + i * 128 * 8), lds_cast(dst + i * 128 * 16), 0, 0);
  }
#endif

  for (int mc = 0; mc < kMC; ++mc) {
#if defined(USE_ASYNC_LDS)
    // Issue next chunk, then wait for the current one (in-order completion),
    // barrier so every wave's slice has landed.
    if (mc + 1 < kMC) {
      const bf16_t* src = vb + (size_t)(mc + 1) * 8192 + (size_t)tid * 8;
      char* dst = vbuf[(mc + 1) & 1] + tid * 16;
      #pragma unroll
      for (int i = 0; i < 8; ++i)
        __builtin_amdgcn_global_load_async_to_lds_b128(
            gsrc_cast(src + i * 128 * 8), lds_cast(dst + i * 128 * 16), 0, 0);
      __builtin_amdgcn_s_wait_asynccnt(8);
    } else {
      __builtin_amdgcn_s_wait_asynccnt(0);
    }
    __syncthreads();
#endif

    // scores: B operand rows kT[m][0..31]; lane col m=l16, K 16*lhalf..+15
    const bf16_t* kr0 = kb + (size_t)(mc * 32 + l16) * kI + lhalf * 16;
    v16bf bk0 = *(const v16bf*)(kr0);
    v16bf bk1 = *(const v16bf*)(kr0 + 16 * kI);
    v8f z = {};
    v8f s0 = wmma_bf16(aq.v, bk0, z);
    v8f s1 = wmma_bf16(aq.v, bk1, z);

    if (mc + 1 < kMC) __builtin_prefetch(kr0 + 32 * kI, 0, 0);

    // online softmax; rows live in VGPR index, cols in lanes (16-lane halves)
    float p0[8], p1[8], scale[8];
    #pragma unroll
    for (int r = 0; r < 8; ++r) {
      float v = fmaxf(s0[r], s1[r]);
      v = fmaxf(v, __shfl_xor(v, 1, 32));
      v = fmaxf(v, __shfl_xor(v, 2, 32));
      v = fmaxf(v, __shfl_xor(v, 4, 32));
      v = fmaxf(v, __shfl_xor(v, 8, 32));
      const float mnew = fmaxf(m_s[r], v);
      scale[r] = __expf(m_s[r] - mnew);
      m_s[r]   = mnew;
      p0[r] = __expf(s0[r] - mnew);
      p1[r] = __expf(s1[r] - mnew);
      float rs = p0[r] + p1[r];
      rs += __shfl_xor(rs, 1, 32);
      rs += __shfl_xor(rs, 2, 32);
      rs += __shfl_xor(rs, 4, 32);
      rs += __shfl_xor(rs, 8, 32);
      l_s[r] = l_s[r] * scale[r] + rs;
    }
    #pragma unroll
    for (int t = 0; t < 16; ++t) {
      #pragma unroll
      for (int r = 0; r < 8; ++r) acc[t][r] *= scale[r];
    }

    // P: D-layout -> LDS [n][m] -> reload as A operand
    #pragma unroll
    for (int r = 0; r < 8; ++r) {
      pstage[w][lhalf * 8 + r][l16]      = f2bf(p0[r]);
      pstage[w][lhalf * 8 + r][16 + l16] = f2bf(p1[r]);
    }
    asm volatile("s_wait_dscnt 0" ::: "memory");
    union { v16bf v; v4u u[2]; } ap;
    {
      const unsigned short* pr = &pstage[w][l16][lhalf * 16];
      ap.u[0] = *(const v4u*)(pr);
      ap.u[1] = *(const v4u*)(pr + 8);
    }

    // O += P * V : 16 c-tiles; V pre-swizzled -> straight 32B/lane operand load
#if defined(USE_ASYNC_LDS)
    const bf16_t* vt = (const bf16_t*)(vbuf[mc & 1]) + lane * 16;
#else
    const bf16_t* vt = vb + ((size_t)mc * 16) * 512 + lane * 16;
#endif
    #pragma unroll
    for (int t = 0; t < 16; ++t) {
      v16bf bv_ = *(const v16bf*)(vt + t * 512);
      acc[t] = wmma_bf16(ap.v, bv_, acc[t]);
    }

#if defined(USE_ASYNC_LDS)
    __syncthreads();   // all reads of this buffer done before it is refilled
#endif
  }

  // epilogue: out = gamma * (O / l) + x   (D layout: col n=l16, rows = c)
  const float g = gamma[0];
  const float* xb = x + (size_t)b * kC * kN;
  float* ob = out + (size_t)b * kC * kN;
  const int n = nbase + l16;
  float inv[8];
  #pragma unroll
  for (int r = 0; r < 8; ++r) inv[r] = 1.0f / l_s[r];
  #pragma unroll
  for (int t = 0; t < 16; ++t) {
    #pragma unroll
    for (int r = 0; r < 8; ++r) {
      const int c = t * 16 + lhalf * 8 + r;
      const size_t idx = (size_t)c * kN + n;
      ob[idx] = g * acc[t][r] * inv[r] + xb[idx];
    }
  }
}

// ---------------------------------------------------------------------------
extern "C" void kernel_launch(void* const* d_in, const int* in_sizes, int n_in,
                              void* d_out, int out_size, void* d_ws, size_t ws_size,
                              hipStream_t stream) {
  (void)in_sizes; (void)n_in; (void)out_size; (void)ws_size;
  const float* x     = (const float*)d_in[0];
  const float* Wq    = (const float*)d_in[1];
  const float* bq    = (const float*)d_in[2];
  const float* Wk    = (const float*)d_in[3];
  const float* bk    = (const float*)d_in[4];
  const float* Wv    = (const float*)d_in[5];
  const float* bv    = (const float*)d_in[6];
  const float* gamma = (const float*)d_in[7];
  float* out = (float*)d_out;

  bf16_t* qT = (bf16_t*)d_ws;                         // 4*4096*32  bf16 = 1 MB
  bf16_t* kT = qT + (size_t)kB * kN * kI;             // 1 MB
  bf16_t* vB = kT + (size_t)kB * kN * kI;             // 8 MB

  proj_kernel<<<dim3(kB * (kN / 32)), dim3(32), 0, stream>>>(
      x, Wq, bq, Wk, bk, Wv, bv, qT, kT, vB);
  attn_kernel<<<dim3(kB * (kN / 16) / 4), dim3(128), 0, stream>>>(
      x, qT, kT, vB, gamma, out);
}
